// SeqPairAttentionOutput_71794673320043
// MI455X (gfx1250) — matrix operand order repaired and checked by reading, not compiled, not run on hardware
//
#include <hip/hip_runtime.h>
#include <cstdint>

typedef __bf16 bf16;
typedef __attribute__((ext_vector_type(16))) __bf16 v16bf;
typedef __attribute__((ext_vector_type(8)))  float   v8f;

// Problem constants (B=1)
constexpr int L  = 512;
constexpr int D  = 1024;
constexpr int H  = 16;
constexpr int HW = 64;
constexpr int N3 = 3 * H * HW;  // 3072
constexpr int DFF = 4 * D;      // 4096

// ------------------------------------------------------------------
// CDNA5 async memory->LDS copy (bypasses VGPRs, tracked by ASYNCcnt)
// ------------------------------------------------------------------
__device__ __forceinline__ void async_copy_b128(uint32_t lds_byte_addr, const void* gptr) {
  asm volatile("global_load_async_to_lds_b128 %0, %1, off"
               :: "v"(lds_byte_addr), "v"(gptr)
               : "memory");
}
__device__ __forceinline__ void wait_async0() {
  asm volatile("s_wait_asynccnt 0" ::: "memory");
}

// ------------------------------------------------------------------
// Generic bf16 WMMA GEMM:  C[M,N] = A[M,K] (row major, bf16)
//                                 @ B[K,N]  (passed as Bt[N,K] row major, bf16)
// f32 accumulate via v_wmma_f32_16x16x32_bf16, fused epilogues.
// Block: 256 threads = 8 waves; tile 128x128, K-step 32 (manually 2x unrolled,
// all K here are multiples of 64). LDS double-buffered; tiles staged with
// global_load_async_to_lds_b128.
// ------------------------------------------------------------------
constexpr int BM = 128, BN = 128, BK = 32;
constexpr int LDSK = 40;               // padded bf16 stride (80B rows, 16B aligned)
constexpr int ABUF = BM * LDSK;        // 5120 bf16
constexpr int TILE = ABUF + BN * LDSK; // 10240 bf16 per buffer (20 KB)

enum { EPI_F32 = 0, EPI_QKV = 1, EPI_SIGMOID = 2, EPI_SCORES = 3,
       EPI_RELU_BF16 = 4, EPI_ADDRES = 5 };

union ABfrag { uint4 u[2]; v16bf v; };

template<int EPI>
__global__ __launch_bounds__(256, 1) void gemm_bf16_wmma(
    const bf16* __restrict__ A, const bf16* __restrict__ Bt,
    int M, int N, int K, int lda, int ldb,
    long long strideA, long long strideB,
    const float* __restrict__ bias,
    const float* __restrict__ add0, long long strideAdd,
    float* __restrict__ outF, bf16* __restrict__ outH,
    int ldc, long long strideC,
    bf16* __restrict__ qout, bf16* __restrict__ kout, bf16* __restrict__ vout)
{
  __shared__ bf16 smem[2 * TILE];  // 40 KB: [buf][A tile | B tile]

  const int tid  = threadIdx.x;
  const int lane = tid & 31;
  const int wave = tid >> 5;
  const int wm   = wave & 3;   // M strip of 32
  const int wn   = wave >> 2;  // N strip of 64
  const int bm   = blockIdx.y * BM;
  const int bn   = blockIdx.x * BN;
  const int batch = blockIdx.z;

  A  += (long long)batch * strideA;
  Bt += (long long)batch * strideB;
  if (outF) outF += (long long)batch * strideC;
  if (outH) outH += (long long)batch * strideC;
  if (add0) add0 += (long long)batch * strideAdd;

  v8f acc[2][4];
  const v8f z8 = {0.f,0.f,0.f,0.f,0.f,0.f,0.f,0.f};
  for (int mi = 0; mi < 2; ++mi)
    for (int ni = 0; ni < 4; ++ni)
      acc[mi][ni] = z8;

  // staging map: thread -> rows r0 / r0+64, 8-element chunk cc (16B)
  const int r0 = tid >> 2;          // 0..63
  const int cc = (tid & 3) * 8;     // 0,8,16,24

  // LDS byte addresses of this thread's 4 staging chunks (within buffer 0)
  const uint32_t smbase = (uint32_t)(uintptr_t)(&smem[0]);
  const uint32_t aoff0 = smbase + (uint32_t)(((r0)      * LDSK + cc) * 2);
  const uint32_t aoff1 = smbase + (uint32_t)(((r0 + 64) * LDSK + cc) * 2);
  const uint32_t boff0 = smbase + (uint32_t)(((BM + r0)      * LDSK + cc) * 2);
  const uint32_t boff1 = smbase + (uint32_t)(((BM + r0 + 64) * LDSK + cc) * 2);
  constexpr uint32_t bufstep = TILE * 2;  // bytes

  // global row pointers (B rows clamped so all async addresses are valid;
  // garbage columns only affect C columns n>=N which are never stored)
  int bn0 = bn + r0;      if (bn0 > N - 1) bn0 = N - 1;
  int bn1 = bn + r0 + 64; if (bn1 > N - 1) bn1 = N - 1;
  const bf16* arow0 = A  + (long long)(bm + r0)      * lda + cc;
  const bf16* arow1 = A  + (long long)(bm + r0 + 64) * lda + cc;
  const bf16* brow0 = Bt + (long long)bn0 * ldb + cc;
  const bf16* brow1 = Bt + (long long)bn1 * ldb + cc;

  // fragment lane maps (ISA 7.12.2, wave32 16-bit layouts)
  const int ml  = lane & 15;
  const int kbA = (lane >> 4) * 8;   // A: K chunks [kbA,kbA+8) and [kbA+16,kbA+24)
  const int kbB = (lane >> 4) * 8;   // B: K chunks [kbB,kbB+8) and [kbB+16,kbB+24)

  auto stage = [&](int kk, uint32_t bufsel) {
    const uint32_t bo = bufsel * bufstep;
    async_copy_b128(aoff0 + bo, arow0 + kk);
    async_copy_b128(aoff1 + bo, arow1 + kk);
    async_copy_b128(boff0 + bo, brow0 + kk);
    async_copy_b128(boff1 + bo, brow1 + kk);
  };

  auto compute = [&](uint32_t bufsel) {
    const bf16* as = smem + bufsel * TILE;
    const bf16* bs = as + ABUF;
    ABfrag a[2], b[4];
#pragma unroll
    for (int mi = 0; mi < 2; ++mi) {
      int row = wm * 32 + mi * 16 + ml;
      a[mi].u[0] = *(const uint4*)&as[row * LDSK + kbA];
      a[mi].u[1] = *(const uint4*)&as[row * LDSK + kbA + 16];
    }
#pragma unroll
    for (int ni = 0; ni < 4; ++ni) {
      int col = wn * 64 + ni * 16 + ml;
      b[ni].u[0] = *(const uint4*)&bs[col * LDSK + kbB];
      b[ni].u[1] = *(const uint4*)&bs[col * LDSK + kbB + 16];
    }
#pragma unroll
    for (int mi = 0; mi < 2; ++mi)
#pragma unroll
      for (int ni = 0; ni < 4; ++ni)
        acc[mi][ni] = __builtin_amdgcn_wmma_f32_16x16x32_bf16(
            false, a[mi].v, false, b[ni].v, (short)0, acc[mi][ni], false, false);
  };

  // prologue: stage tile 0 into buffer 0
  stage(0, 0);

  // K is always a multiple of 2*BK here -> clean 2x unrolled pipeline.
  for (int k0 = 0; k0 < K; k0 += 2 * BK) {
    wait_async0();        // own copies for buffer 0 complete
    __syncthreads();      // whole tile resident for all waves
    if (k0 + BK < K) stage(k0 + BK, 1);
    compute(0);

    wait_async0();        // own copies for buffer 1 complete
    __syncthreads();
    if (k0 + 2 * BK < K) stage(k0 + 2 * BK, 0);
    compute(1);
    // frag ds_loads are consumed (dscnt-waited) before the next barrier, and
    // async writes always target the buffer not being read -> one barrier per
    // K-step is sufficient.
  }

  // epilogue: C/D frag layout — VGPR j holds M = j + 8*(lane>>4), N = lane&15
  const int mo = (lane >> 4) * 8;
  const int nl = lane & 15;
#pragma unroll
  for (int mi = 0; mi < 2; ++mi)
#pragma unroll
    for (int ni = 0; ni < 4; ++ni)
#pragma unroll
      for (int j = 0; j < 8; ++j) {
        int m = bm + wm * 32 + mi * 16 + mo + j;
        int n = bn + wn * 64 + ni * 16 + nl;
        if (m >= M || n >= N) continue;
        float v = acc[mi][ni][j];
        if (EPI == EPI_F32) {
          outF[(long long)m * ldc + n] = v;
        } else if (EPI == EPI_QKV) {
          int h = n / 192, r = n - h * 192;
          int comp = r >> 6, c = r & 63;
          if (comp == 0)       qout[((long long)h * L + m) * HW + c] = (bf16)(v * 0.125f);
          else if (comp == 1)  kout[((long long)h * L + m) * HW + c] = (bf16)v;
          else                 vout[((long long)h * HW + c) * L + m] = (bf16)v;  // v transposed
        } else if (EPI == EPI_SIGMOID) {
          float t = v + bias[n];
          outF[(long long)m * ldc + n] = 1.f / (1.f + __expf(-t));
        } else if (EPI == EPI_SCORES) {
          long long idx = (long long)m * ldc + n;
          outF[idx] = v + add0[idx];   // in-place: scores = qk + pair bias
        } else if (EPI == EPI_RELU_BF16) {
          float t = v + bias[n];
          outH[(long long)m * ldc + n] = (bf16)fmaxf(t, 0.f);
        } else if (EPI == EPI_ADDRES) {
          outF[(long long)m * ldc + n] = v + bias[n] + add0[(long long)m * ldc + n];
        }
      }
}

// ------------------------------------------------------------------
// fp32 [K,N] -> bf16 [N,K] convert + transpose (weights, once per call)
// ------------------------------------------------------------------
__global__ __launch_bounds__(256) void convert_transpose_kernel(
    const float* __restrict__ in, bf16* __restrict__ out, int K, int N)
{
  __shared__ float t[32][33];
  int bx = blockIdx.x * 32;  // N
  int by = blockIdx.y * 32;  // K
  int tx = threadIdx.x & 31, ty = threadIdx.x >> 5;
  for (int i = 0; i < 32; i += 8) {
    int k = by + ty + i, n = bx + tx;
    t[ty + i][tx] = (k < K && n < N) ? in[(long long)k * N + n] : 0.f;
  }
  __syncthreads();
  for (int i = 0; i < 32; i += 8) {
    int n = bx + ty + i, k = by + tx;
    if (n < N && k < K) out[(long long)n * K + k] = (bf16)t[tx][ty + i];
  }
}

// ------------------------------------------------------------------
// Pair bias: LN over PD=128 then project 128->16.  One wave per (q,k).
// bias layout: [H, L, L] fp32 (becomes the scores buffer).
// ------------------------------------------------------------------
__global__ __launch_bounds__(256) void pair_bias_kernel(
    const float* __restrict__ z, const float* __restrict__ g,
    const float* __restrict__ b, const float* __restrict__ w,
    float* __restrict__ bias)
{
  __shared__ float wsm[128 * 16];
  for (int i = threadIdx.x; i < 128 * 16; i += 256) wsm[i] = w[i];
  __syncthreads();

  const int lane = threadIdx.x & 31;
  const int wave = threadIdx.x >> 5;
  const long long pair = (long long)blockIdx.x * 8 + wave;  // 0..262143

  float4 v4 = *(const float4*)(z + pair * 128 + lane * 4);
  float zv[4] = {v4.x, v4.y, v4.z, v4.w};
  float s = zv[0] + zv[1] + zv[2] + zv[3];
  float ss = zv[0]*zv[0] + zv[1]*zv[1] + zv[2]*zv[2] + zv[3]*zv[3];
  for (int o = 16; o; o >>= 1) { s += __shfl_xor(s, o); ss += __shfl_xor(ss, o); }
  float mu = s * (1.f / 128.f);
  float var = ss * (1.f / 128.f) - mu * mu;
  float rs = rsqrtf(var + 1e-5f);

  const int p0 = lane * 4;
  float acc[16];
  for (int h = 0; h < 16; ++h) acc[h] = 0.f;
  for (int j = 0; j < 4; ++j) {
    float zn = (zv[j] - mu) * rs * g[p0 + j] + b[p0 + j];
    const float* wr = &wsm[(p0 + j) * 16];
    for (int h = 0; h < 16; ++h) acc[h] += zn * wr[h];
  }
  for (int o = 16; o; o >>= 1)
    for (int h = 0; h < 16; ++h) acc[h] += __shfl_xor(acc[h], o);
  if (lane < 16) bias[(long long)lane * (L * L) + pair] = acc[lane];
}

// ------------------------------------------------------------------
// Row LayerNorm over 1024 cols -> bf16. One 256-thread block per row.
// ------------------------------------------------------------------
__global__ __launch_bounds__(256) void ln_rows_kernel(
    const float* __restrict__ x, const float* __restrict__ g,
    const float* __restrict__ b, bf16* __restrict__ y)
{
  __shared__ float red[16];
  const int row = blockIdx.x;
  const int lane = threadIdx.x & 31, wave = threadIdx.x >> 5;
  float4 v = ((const float4*)(x + (long long)row * D))[threadIdx.x];
  float s = v.x + v.y + v.z + v.w;
  float ss = v.x*v.x + v.y*v.y + v.z*v.z + v.w*v.w;
  for (int o = 16; o; o >>= 1) { s += __shfl_xor(s, o); ss += __shfl_xor(ss, o); }
  if (lane == 0) { red[wave] = s; red[wave + 8] = ss; }
  __syncthreads();
  if (wave == 0) {
    float s2  = (lane < 8) ? red[lane] : 0.f;
    float ss2 = (lane < 8) ? red[lane + 8] : 0.f;
    for (int o = 4; o; o >>= 1) { s2 += __shfl_xor(s2, o); ss2 += __shfl_xor(ss2, o); }
    if (lane == 0) { red[0] = s2; red[1] = ss2; }
  }
  __syncthreads();
  float mu = red[0] * (1.f / (float)D);
  float var = red[1] * (1.f / (float)D) - mu * mu;
  float rs = rsqrtf(var + 1e-5f);
  int c = threadIdx.x * 4;
  union { bf16 h[4]; uint2 u; } pk;
  pk.h[0] = (bf16)((v.x - mu) * rs * g[c + 0] + b[c + 0]);
  pk.h[1] = (bf16)((v.y - mu) * rs * g[c + 1] + b[c + 1]);
  pk.h[2] = (bf16)((v.z - mu) * rs * g[c + 2] + b[c + 2]);
  pk.h[3] = (bf16)((v.w - mu) * rs * g[c + 3] + b[c + 3]);
  ((uint2*)y)[(long long)row * (D / 4) + threadIdx.x] = pk.u;
}

// ------------------------------------------------------------------
// Softmax over 512-wide rows with key mask; fp32 scores -> bf16 P.
// One wave per row (16 elements/lane). rows = H*L = 8192.
// ------------------------------------------------------------------
__global__ __launch_bounds__(256) void softmax_kernel(
    const float* __restrict__ scores, const unsigned char* __restrict__ mask,
    bf16* __restrict__ P)
{
  const int lane = threadIdx.x & 31, wave = threadIdx.x >> 5;
  const long long row = (long long)blockIdx.x * 8 + wave;
  union { float4 f4[4]; float f[16]; } u;
  const float4* s4 = (const float4*)(scores + row * L + lane * 16);
  u.f4[0] = s4[0]; u.f4[1] = s4[1]; u.f4[2] = s4[2]; u.f4[3] = s4[3];
  const int kbase = lane * 16;
  for (int j = 0; j < 16; ++j)
    if (!mask[kbase + j]) u.f[j] = -INFINITY;
  float mx = u.f[0];
  for (int j = 1; j < 16; ++j) mx = fmaxf(mx, u.f[j]);
  for (int o = 16; o; o >>= 1) mx = fmaxf(mx, __shfl_xor(mx, o));
  float e[16], sum = 0.f;
  for (int j = 0; j < 16; ++j) { e[j] = __expf(u.f[j] - mx); sum += e[j]; }
  for (int o = 16; o; o >>= 1) sum += __shfl_xor(sum, o);
  float inv = 1.f / sum;
  union { bf16 h[16]; uint4 q[2]; } pk;
  for (int j = 0; j < 16; ++j) pk.h[j] = (bf16)(e[j] * inv);
  uint4* dst = (uint4*)(P + row * L + lane * 16);
  dst[0] = pk.q[0]; dst[1] = pk.q[1];
}

// o_bf = bf16(gate * o), elementwise over 512*1024
__global__ __launch_bounds__(256) void gatemul_kernel(
    const float* __restrict__ g, const float* __restrict__ o, bf16* __restrict__ out)
{
  int i = blockIdx.x * 256 + threadIdx.x;
  out[i] = (bf16)(g[i] * o[i]);
}

// ------------------------------------------------------------------
extern "C" void kernel_launch(void* const* d_in, const int* in_sizes, int n_in,
                              void* d_out, int out_size, void* d_ws, size_t ws_size,
                              hipStream_t stream) {
  (void)in_sizes; (void)n_in; (void)out_size; (void)ws_size;
  const float* seq      = (const float*)d_in[0];   // [L,D]
  const float* pairwise = (const float*)d_in[1];   // [L,L,128]
  const unsigned char* mask = (const unsigned char*)d_in[2];  // [L] bool
  const float* ln1_g = (const float*)d_in[3];
  const float* ln1_b = (const float*)d_in[4];
  const float* proj_w = (const float*)d_in[5];     // [D, 3072]
  const float* g_w   = (const float*)d_in[6];      // [D, D]
  const float* g_b   = (const float*)d_in[7];
  const float* o_w   = (const float*)d_in[8];      // [D, D]
  const float* o_b   = (const float*)d_in[9];
  const float* p2s_ln_g = (const float*)d_in[10];
  const float* p2s_ln_b = (const float*)d_in[11];
  const float* p2s_w    = (const float*)d_in[12];  // [128,16]
  const float* mlp_ln_g = (const float*)d_in[13];
  const float* mlp_ln_b = (const float*)d_in[14];
  const float* mlp_w1   = (const float*)d_in[15];  // [D, 4096]
  const float* mlp_b1   = (const float*)d_in[16];
  const float* mlp_w2   = (const float*)d_in[17];  // [4096, D]
  const float* mlp_b2   = (const float*)d_in[18];
  float* out = (float*)d_out;

  // workspace carve-up (256B aligned)
  char* ws = (char*)d_ws;
  size_t off = 0;
  auto alloc = [&](size_t bytes) { size_t o = off; off = (off + bytes + 255) & ~(size_t)255; return o; };
  float* scores = (float*)(ws + alloc((size_t)H * L * L * 4));  // pair bias, then scores (in-place)
  bf16*  Pm     = (bf16*) (ws + alloc((size_t)H * L * L * 2));
  bf16*  ybf    = (bf16*) (ws + alloc((size_t)L * D * 2));
  bf16*  qbf    = (bf16*) (ws + alloc((size_t)H * L * HW * 2));
  bf16*  kbf    = (bf16*) (ws + alloc((size_t)H * L * HW * 2));
  bf16*  vtbf   = (bf16*) (ws + alloc((size_t)H * HW * L * 2));
  float* gate   = (float*)(ws + alloc((size_t)L * D * 4));
  float* oacc   = (float*)(ws + alloc((size_t)L * D * 4));
  bf16*  obf    = (bf16*) (ws + alloc((size_t)L * D * 2));
  float* sres   = (float*)(ws + alloc((size_t)L * D * 4));
  bf16*  hbf    = (bf16*) (ws + alloc((size_t)L * D * 2));
  bf16*  m1     = (bf16*) (ws + alloc((size_t)L * DFF * 2));
  bf16*  projWt = (bf16*) (ws + alloc((size_t)N3 * D * 2));
  bf16*  gWt    = (bf16*) (ws + alloc((size_t)D * D * 2));
  bf16*  oWt    = (bf16*) (ws + alloc((size_t)D * D * 2));
  bf16*  w1t    = (bf16*) (ws + alloc((size_t)DFF * D * 2));
  bf16*  w2t    = (bf16*) (ws + alloc((size_t)D * DFF * 2));

  // 1) weight convert + transpose to bf16 [N,K]
  convert_transpose_kernel<<<dim3(N3/32,  D/32),  256, 0, stream>>>(proj_w, projWt, D,   N3);
  convert_transpose_kernel<<<dim3(D/32,   D/32),  256, 0, stream>>>(g_w,    gWt,    D,   D);
  convert_transpose_kernel<<<dim3(D/32,   D/32),  256, 0, stream>>>(o_w,    oWt,    D,   D);
  convert_transpose_kernel<<<dim3(DFF/32, D/32),  256, 0, stream>>>(mlp_w1, w1t,    D,   DFF);
  convert_transpose_kernel<<<dim3(D/32,   DFF/32),256, 0, stream>>>(mlp_w2, w2t,    DFF, D);

  // 2) pair bias -> scores buffer [H,L,L]
  pair_bias_kernel<<<(L * L) / 8, 256, 0, stream>>>(pairwise, p2s_ln_g, p2s_ln_b, p2s_w, scores);

  // 3) pre-attention LN -> y (bf16)
  ln_rows_kernel<<<L, 256, 0, stream>>>(seq, ln1_g, ln1_b, ybf);

  // 4) QKV projection: [512,1024]x[1024,3072]; scatter q(scaled)/k/v(transposed)
  gemm_bf16_wmma<EPI_QKV><<<dim3(N3/BN, L/BM, 1), 256, 0, stream>>>(
      ybf, projWt, L, N3, D, D, D, 0, 0,
      nullptr, nullptr, 0, nullptr, nullptr, 0, 0, qbf, kbf, vtbf);

  // 5) gate = sigmoid(y @ g_w + g_b)
  gemm_bf16_wmma<EPI_SIGMOID><<<dim3(D/BN, L/BM, 1), 256, 0, stream>>>(
      ybf, gWt, L, D, D, D, D, 0, 0,
      g_b, nullptr, 0, gate, nullptr, D, 0, nullptr, nullptr, nullptr);

  // 6) scores[h] = q[h] @ k[h]^T + pair_bias[h]   (batched over heads, in-place)
  gemm_bf16_wmma<EPI_SCORES><<<dim3(L/BN, L/BM, H), 256, 0, stream>>>(
      qbf, kbf, L, L, HW, HW, HW, (long long)L * HW, (long long)L * HW,
      nullptr, scores, (long long)L * L, scores, nullptr, L, (long long)L * L,
      nullptr, nullptr, nullptr);

  // 7) softmax rows (with key mask) -> P (bf16)
  softmax_kernel<<<(H * L) / 8, 256, 0, stream>>>(scores, mask, Pm);

  // 8) o[h] = P[h] @ v[h] -> packed [m, h*64+c] fp32
  gemm_bf16_wmma<EPI_F32><<<dim3(1, L/BM, H), 256, 0, stream>>>(
      Pm, vtbf, L, HW, L, L, L, (long long)L * L, (long long)HW * L,
      nullptr, nullptr, 0, oacc, nullptr, D, (long long)HW,
      nullptr, nullptr, nullptr);

  // 9) gated output (bf16)
  gatemul_kernel<<<(L * D) / 256, 256, 0, stream>>>(gate, oacc, obf);

  // 10) s = seq + (gate*o) @ o_w + o_b
  gemm_bf16_wmma<EPI_ADDRES><<<dim3(D/BN, L/BM, 1), 256, 0, stream>>>(
      obf, oWt, L, D, D, D, D, 0, 0,
      o_b, seq, 0, sres, nullptr, D, 0, nullptr, nullptr, nullptr);

  // 11) MLP LN -> h (bf16)
  ln_rows_kernel<<<L, 256, 0, stream>>>(sres, mlp_ln_g, mlp_ln_b, hbf);

  // 12) m1 = relu(h @ w1 + b1) (bf16)
  gemm_bf16_wmma<EPI_RELU_BF16><<<dim3(DFF/BN, L/BM, 1), 256, 0, stream>>>(
      hbf, w1t, L, DFF, D, D, D, 0, 0,
      mlp_b1, nullptr, 0, nullptr, m1, DFF, 0, nullptr, nullptr, nullptr);

  // 13) out = s + m1 @ w2 + b2
  gemm_bf16_wmma<EPI_ADDRES><<<dim3(D/BN, L/BM, 1), 256, 0, stream>>>(
      m1, w2t, L, D, DFF, DFF, DFF, 0, 0,
      mlp_b2, sres, 0, out, nullptr, D, 0, nullptr, nullptr, nullptr);
}
